// GRUBlock_37787122270913
// MI455X (gfx1250) — matrix operand order repaired and checked by reading, not compiled, not run on hardware
//
#include <hip/hip_runtime.h>
#include <hip/hip_bf16.h>

typedef __attribute__((ext_vector_type(16))) __bf16 v16bf;
typedef __attribute__((ext_vector_type(8)))  float  v8f;

#define B_DIM 64
#define T_DIM 512
#define I_DIM 1024
#define H_DIM 1024
#define BT    (B_DIM * T_DIM)       // 32768
#define H3    (3 * H_DIM)           // 3072

// ---------------- WMMA fragment helpers (bf16 16x16x32, f32 acc) ----------------
// A fragment: 16(M) x 32(K): lane<16 -> row M=lane, K {0..7,16..23}; lane>=16 -> K {8..15,24..31}
static __device__ __forceinline__ v16bf load_frag_a(const __hip_bfloat16* p0, int ld, int lane) {
    const int r = lane & 15, hh = lane >> 4;
    const __hip_bfloat16* p = p0 + (size_t)r * ld + hh * 8;
    union { v16bf v; uint4 u[2]; } f;
    f.u[0] = *(const uint4*)(p);
    f.u[1] = *(const uint4*)(p + 16);
    return f.v;
}

// B fragment: 32(K) x 16(N) from [N,K] row-major: lane<16 -> col N=lane, K {0..7,16..23}; lane>=16 -> K {8..15,24..31}
static __device__ __forceinline__ v16bf load_frag_b(const __hip_bfloat16* p0, int ld, int lane) {
    const int r = lane & 15, hh = lane >> 4;
    const __hip_bfloat16* p = p0 + (size_t)r * ld + hh * 8;
    union { v16bf v; uint4 u[2]; } f;
    f.u[0] = *(const uint4*)(p);
    f.u[1] = *(const uint4*)(p + 16);
    return f.v;
}

static __device__ __forceinline__ v8f wmma_bf16(v16bf a, v16bf b, v8f c) {
    return __builtin_amdgcn_wmma_f32_16x16x32_bf16(false, a, false, b, (short)0, c, false, false);
}

static __device__ __forceinline__ float sigmoidf_fast(float x) {
    return 1.0f / (1.0f + __expf(-x));
}

// Async global->LDS copy of 16 bytes per lane (CDNA5 ASYNCcnt path).
static __device__ __forceinline__ void async_copy_b128(unsigned lds_off, const __hip_bfloat16* gptr) {
    asm volatile("global_load_async_to_lds_b128 %0, %1, off"
                 :: "v"(lds_off), "v"(gptr)
                 : "memory");
}

// ---------------- elementwise utility kernels ----------------
__global__ void f32_to_bf16_kernel(const float* __restrict__ in,
                                   __hip_bfloat16* __restrict__ out, int n) {
    int i = blockIdx.x * blockDim.x + threadIdx.x;
    if (i < n) out[i] = __float2bfloat16(in[i]);
}

__global__ void zero_u32_kernel(unsigned int* __restrict__ p, int n) {
    int i = blockIdx.x * blockDim.x + threadIdx.x;
    if (i < n) p[i] = 0u;
}

// ---------------- GEMM: C[M,N] = A[M,K] * W[N,K]^T, async-LDS double-buffered ----------------
// 256 threads = 8 waves (2 M x 4 N); block tile 64x128, K chunk 64; 48 KB LDS ping-pong.
// Per stage: 6 async b128 copies/thread, 8 WMMAs + 16 ds_load_b128 per wave, 2 barriers.
template <bool OUT_BF16>
__global__ void wmma_gemm_async_kernel(const __hip_bfloat16* __restrict__ A,
                                       const __hip_bfloat16* __restrict__ W,
                                       void* __restrict__ C,
                                       int M, int N, int K) {
    __shared__ alignas(16) __hip_bfloat16 Abuf[2][64 * 64];   // 2 x 8 KB
    __shared__ alignas(16) __hip_bfloat16 Bbuf[2][128 * 64];  // 2 x 16 KB

    const int tid  = threadIdx.x;
    const int lane = tid & 31;
    const int wave = tid >> 5;
    const int wm = wave >> 2;                 // 0..1 -> 32 rows
    const int wn = wave & 3;                  // 0..3 -> 32 cols
    const int row0 = blockIdx.y * 64;
    const int col0 = blockIdx.x * 128;

    // copy indices: tile rows are 64 elems = 8 chunks of 16B; 256 threads move
    // 2 chunks of A (64x64) and 4 chunks of B (128x64) per stage.
    const int cr = tid >> 3;                  // 0..31 base row
    const int cc = (tid & 7) * 8;             // element offset within 64-wide row

    v8f acc00 = {}, acc01 = {}, acc10 = {}, acc11 = {};

    auto issue_stage = [&](int buf, int k0) {
        // A tile: rows cr and cr+32
        async_copy_b128((unsigned)(size_t)&Abuf[buf][cr * 64 + cc],
                        A + (size_t)(row0 + cr) * K + k0 + cc);
        async_copy_b128((unsigned)(size_t)&Abuf[buf][(cr + 32) * 64 + cc],
                        A + (size_t)(row0 + cr + 32) * K + k0 + cc);
        // B tile: rows cr, cr+32, cr+64, cr+96
#pragma unroll
        for (int s = 0; s < 4; ++s) {
            async_copy_b128((unsigned)(size_t)&Bbuf[buf][(cr + s * 32) * 64 + cc],
                            W + (size_t)(col0 + cr + s * 32) * K + k0 + cc);
        }
    };

    const int NK = K >> 6;                    // K chunks of 64
    issue_stage(0, 0);
    for (int kc = 0; kc < NK; ++kc) {
        const int cur = kc & 1;
        if (kc + 1 < NK) {
            issue_stage(cur ^ 1, (kc + 1) << 6);
            asm volatile("s_wait_asynccnt 0x6" ::: "memory");  // previous stage landed
        } else {
            asm volatile("s_wait_asynccnt 0x0" ::: "memory");
        }
        __syncthreads();

        const __hip_bfloat16* At = &Abuf[cur][0];
        const __hip_bfloat16* Bt = &Bbuf[cur][0];
#pragma unroll
        for (int ks = 0; ks < 64; ks += 32) {
            v16bf a0 = load_frag_a(At + (wm * 32 + 0)  * 64 + ks, 64, lane);
            v16bf a1 = load_frag_a(At + (wm * 32 + 16) * 64 + ks, 64, lane);
            v16bf b0 = load_frag_b(Bt + (wn * 32 + 0)  * 64 + ks, 64, lane);
            v16bf b1 = load_frag_b(Bt + (wn * 32 + 16) * 64 + ks, 64, lane);
            acc00 = wmma_bf16(a0, b0, acc00);
            acc01 = wmma_bf16(a0, b1, acc01);
            acc10 = wmma_bf16(a1, b0, acc10);
            acc11 = wmma_bf16(a1, b1, acc11);
        }

        __syncthreads();   // protect buffer before it is refilled
    }

    // C layout: VGPR i, lanes<16 -> M=i, lanes>=16 -> M=i+8; N = lane&15
    const int mlo = (lane >> 4) * 8;
    const int nn  = lane & 15;
    const int wr0 = row0 + wm * 32;
    const int wc0 = col0 + wn * 32;
#pragma unroll
    for (int i = 0; i < 8; ++i) {
        const int r0 = wr0 + mlo + i;
        const int r1 = r0 + 16;
        const int c0 = wc0 + nn;
        const int c1 = c0 + 16;
        if (OUT_BF16) {
            __hip_bfloat16* Cb = (__hip_bfloat16*)C;
            Cb[(size_t)r0 * N + c0] = __float2bfloat16(acc00[i]);
            Cb[(size_t)r0 * N + c1] = __float2bfloat16(acc01[i]);
            Cb[(size_t)r1 * N + c0] = __float2bfloat16(acc10[i]);
            Cb[(size_t)r1 * N + c1] = __float2bfloat16(acc11[i]);
        } else {
            float* Cf = (float*)C;
            Cf[(size_t)r0 * N + c0] = acc00[i];
            Cf[(size_t)r0 * N + c1] = acc01[i];
            Cf[(size_t)r1 * N + c0] = acc10[i];
            Cf[(size_t)r1 * N + c1] = acc11[i];
        }
    }
}

// ---------------- fused GRU step (one launch per timestep) ----------------
// grid.x = H/64 = 16 blocks; 8 waves (4 M x 2 N); wave = 16(batch) x 32(hidden); K = H.
__global__ void gru_step_kernel(const __hip_bfloat16* __restrict__ h_bf,
                                const float* __restrict__ h_f32,
                                const __hip_bfloat16* __restrict__ w_hh,
                                const __hip_bfloat16* __restrict__ gx,
                                __hip_bfloat16* __restrict__ hs,
                                __hip_bfloat16* __restrict__ h_bf_out,
                                float* __restrict__ h_f32_out,
                                int t) {
    const int lane = threadIdx.x & 31;
    const int wave = threadIdx.x >> 5;
    const int wm = wave >> 1;
    const int wn = wave & 1;
    const int row0 = wm * 16;
    const int col0 = blockIdx.x * 64 + wn * 32;

    v8f acc[3][2] = {};
    const __hip_bfloat16* Arow = h_bf + (size_t)row0 * H_DIM;
    const __hip_bfloat16* W0 = w_hh + (size_t)(0 * H_DIM + col0) * H_DIM;
    const __hip_bfloat16* W1 = w_hh + (size_t)(1 * H_DIM + col0) * H_DIM;
    const __hip_bfloat16* W2 = w_hh + (size_t)(2 * H_DIM + col0) * H_DIM;

    for (int k0 = 0; k0 < H_DIM; k0 += 32) {
        // load all fragments for this chunk into distinct registers first
        v16bf a   = load_frag_a(Arow + k0, H_DIM, lane);
        v16bf b00 = load_frag_b(W0 + k0, H_DIM, lane);
        v16bf b01 = load_frag_b(W0 + (size_t)16 * H_DIM + k0, H_DIM, lane);
        v16bf b10 = load_frag_b(W1 + k0, H_DIM, lane);
        v16bf b11 = load_frag_b(W1 + (size_t)16 * H_DIM + k0, H_DIM, lane);
        v16bf b20 = load_frag_b(W2 + k0, H_DIM, lane);
        v16bf b21 = load_frag_b(W2 + (size_t)16 * H_DIM + k0, H_DIM, lane);
        acc[0][0] = wmma_bf16(a, b00, acc[0][0]);
        acc[0][1] = wmma_bf16(a, b01, acc[0][1]);
        acc[1][0] = wmma_bf16(a, b10, acc[1][0]);
        acc[1][1] = wmma_bf16(a, b11, acc[1][1]);
        acc[2][0] = wmma_bf16(a, b20, acc[2][0]);
        acc[2][1] = wmma_bf16(a, b21, acc[2][1]);
    }

    const int mlo = (lane >> 4) * 8;
    const int nn  = lane & 15;
#pragma unroll
    for (int i = 0; i < 8; ++i) {
#pragma unroll
        for (int j = 0; j < 2; ++j) {
            const int b = row0 + mlo + i;
            const int c = col0 + j * 16 + nn;
            const size_t gxbase = ((size_t)b * T_DIM + t) * H3 + c;

            const float xr = __bfloat162float(gx[gxbase]);
            const float xz = __bfloat162float(gx[gxbase + H_DIM]);
            const float xn = __bfloat162float(gx[gxbase + 2 * H_DIM]);

            const float r  = sigmoidf_fast(xr + acc[0][j][i]);
            const float z  = sigmoidf_fast(xz + acc[1][j][i]);
            const float nv = tanhf(xn + r * acc[2][j][i]);

            const float hp   = h_f32[(size_t)b * H_DIM + c];
            const float hnew = (1.0f - z) * nv + z * hp;

            h_f32_out[(size_t)b * H_DIM + c] = hnew;
            const __hip_bfloat16 hb = __float2bfloat16(hnew);
            h_bf_out[(size_t)b * H_DIM + c] = hb;
            hs[((size_t)b * T_DIM + t) * H_DIM + c] = hb;
        }
    }
}

// ---------------- workspace layout (bytes) ----------------
static constexpr size_t OFF_XBF    = 0;
static constexpr size_t OFF_WIH    = OFF_XBF   + (size_t)BT * I_DIM * 2;
static constexpr size_t OFF_WHH    = OFF_WIH   + (size_t)H3 * I_DIM * 2;
static constexpr size_t OFF_WPROJ  = OFF_WHH   + (size_t)H3 * H_DIM * 2;
static constexpr size_t OFF_GX     = OFF_WPROJ + (size_t)H_DIM * H_DIM * 2;
static constexpr size_t OFF_HS     = OFF_GX    + (size_t)BT * H3 * 2;
static constexpr size_t OFF_HF32   = OFF_HS    + (size_t)BT * H_DIM * 2;
static constexpr size_t OFF_HBF    = OFF_HF32  + (size_t)2 * B_DIM * H_DIM * 4;

extern "C" void kernel_launch(void* const* d_in, const int* in_sizes, int n_in,
                              void* d_out, int out_size, void* d_ws, size_t ws_size,
                              hipStream_t stream) {
    const float* x      = (const float*)d_in[0];
    const float* w_ih   = (const float*)d_in[1];
    const float* w_hh   = (const float*)d_in[2];
    const float* w_proj = (const float*)d_in[3];
    float* out = (float*)d_out;

    char* ws = (char*)d_ws;
    __hip_bfloat16* x_bf    = (__hip_bfloat16*)(ws + OFF_XBF);
    __hip_bfloat16* w_ih_bf = (__hip_bfloat16*)(ws + OFF_WIH);
    __hip_bfloat16* w_hh_bf = (__hip_bfloat16*)(ws + OFF_WHH);
    __hip_bfloat16* w_pj_bf = (__hip_bfloat16*)(ws + OFF_WPROJ);
    __hip_bfloat16* gx_bf   = (__hip_bfloat16*)(ws + OFF_GX);
    __hip_bfloat16* hs_bf   = (__hip_bfloat16*)(ws + OFF_HS);
    float*          h_f32   = (float*)(ws + OFF_HF32);
    __hip_bfloat16* h_bf    = (__hip_bfloat16*)(ws + OFF_HBF);

    // 1) convert inputs to bf16
    {
        int n;
        n = BT * I_DIM;
        f32_to_bf16_kernel<<<(n + 255) / 256, 256, 0, stream>>>(x, x_bf, n);
        n = H3 * I_DIM;
        f32_to_bf16_kernel<<<(n + 255) / 256, 256, 0, stream>>>(w_ih, w_ih_bf, n);
        n = H3 * H_DIM;
        f32_to_bf16_kernel<<<(n + 255) / 256, 256, 0, stream>>>(w_hh, w_hh_bf, n);
        n = H_DIM * H_DIM;
        f32_to_bf16_kernel<<<(n + 255) / 256, 256, 0, stream>>>(w_proj, w_pj_bf, n);
    }

    // 2) zero-init h ping-pong buffers (f32 + bf16 regions contiguous)
    {
        int ndw = (2 * B_DIM * H_DIM * 4 + 2 * B_DIM * H_DIM * 2) / 4;
        zero_u32_kernel<<<(ndw + 255) / 256, 256, 0, stream>>>(
            (unsigned int*)(ws + OFF_HF32), ndw);
    }

    // 3) gx = x * w_ih^T -> bf16 [32768,3072]
    wmma_gemm_async_kernel<true><<<dim3(H3 / 128, BT / 64), 256, 0, stream>>>(
        x_bf, w_ih_bf, gx_bf, BT, H3, I_DIM);

    // 4) sequential GRU recurrence: 512 fused step launches (ping-pong h)
    for (int t = 0; t < T_DIM; ++t) {
        const int pin = t & 1, pout = (t + 1) & 1;
        gru_step_kernel<<<dim3(H_DIM / 64), 256, 0, stream>>>(
            h_bf + (size_t)pin * B_DIM * H_DIM,
            h_f32 + (size_t)pin * B_DIM * H_DIM,
            w_hh_bf, gx_bf, hs_bf,
            h_bf + (size_t)pout * B_DIM * H_DIM,
            h_f32 + (size_t)pout * B_DIM * H_DIM,
            t);
    }

    // 5) out = hs * w_proj^T -> f32 [32768,1024]
    wmma_gemm_async_kernel<false><<<dim3(H_DIM / 128, BT / 64), 256, 0, stream>>>(
        hs_bf, w_pj_bf, out, BT, H_DIM, H_DIM);
}
